// GroupQueryAttention_67671504716070
// MI455X (gfx1250) — matrix operand order repaired and checked by reading, not compiled, not run on hardware
//
#include <hip/hip_runtime.h>
#include <hip/hip_bf16.h>

// ---------------- problem constants ----------------
constexpr int B   = 2;
constexpr int S   = 2048;
constexpr int H   = 2048;
constexpr int NH  = 16;
constexpr int KVH = 4;
constexpr int D   = 128;
constexpr int G   = NH / KVH;     // 4
constexpr int KVD = KVH * D;      // 512
constexpr int M_TOK = B * S;      // 4096 rows

typedef __bf16 bf16_t;
typedef __attribute__((ext_vector_type(16))) __bf16 v16bf;
typedef __attribute__((ext_vector_type(8)))  __bf16 v8bf;
typedef __attribute__((ext_vector_type(8)))  float  v8f;

#define NEG_INF (-3.0e38f)

// ---------------- fragment helpers (contiguous k order, same for A and B) ----------------
// A-matrix 16x32 bf16: lane r (0-15) = row r, halves 0..15 = K0..15
//                      lane r+16     = row r, halves 0..15 = K16..31
static __device__ inline v16bf load_a_frag(const bf16_t* rowPtr, int kbase, int hi) {
  v8bf c0 = *(const v8bf*)(rowPtr + kbase + hi * 16);
  v8bf c1 = *(const v8bf*)(rowPtr + kbase + hi * 16 + 8);
  v16bf a;
#pragma unroll
  for (int i = 0; i < 8; ++i) { a[i] = c0[i]; a[i + 8] = c1[i]; }
  return a;
}

static __device__ inline v8f wmma_bf16(v16bf a, v16bf b, v8f c) {
  return __builtin_amdgcn_wmma_f32_16x16x32_bf16(false, a, false, b, (short)0, c,
                                                 false, false);
}

// ---------------- elementwise fp32 -> bf16 ----------------
__global__ void cvt_f32_bf16(const float* __restrict__ in, bf16_t* __restrict__ out, int n) {
  int i = blockIdx.x * blockDim.x + threadIdx.x;
  if (i < n) out[i] = (bf16_t)in[i];
}

// ---------------- wave-tile GEMM: C[M,N] = A[M,K] * W[N,K]^T + bias ----------------
// one wave32 per 32x64 register tile: 2 A-frags x 4 B-frags -> 8 wmma per K-step.
// ~21 FLOP/byte from L2 vs ~5 for a 16x16 tile: keeps the matrix pipes fed.
__global__ void __launch_bounds__(32)
gemm_bf16_wave(const bf16_t* __restrict__ A, const bf16_t* __restrict__ W,
               const float* __restrict__ bias, bf16_t* __restrict__ Cb,
               float* __restrict__ Cf, int M, int N, int K) {
  const int lane = threadIdx.x;
  const int r  = lane & 15;
  const int hi = lane >> 4;
  const int m0 = blockIdx.y * 32;
  const int n0 = blockIdx.x * 64;

  v8f acc[2][4];
#pragma unroll
  for (int mi = 0; mi < 2; ++mi)
#pragma unroll
    for (int t = 0; t < 4; ++t)
#pragma unroll
      for (int i = 0; i < 8; ++i) acc[mi][t][i] = 0.0f;

  const bf16_t* arow0 = A + (size_t)(m0 + r) * K;
  const bf16_t* arow1 = A + (size_t)(m0 + 16 + r) * K;
  const bf16_t* wrow0 = W + (size_t)(n0 + r) * K;
  const bf16_t* wrow1 = W + (size_t)(n0 + 16 + r) * K;
  const bf16_t* wrow2 = W + (size_t)(n0 + 32 + r) * K;
  const bf16_t* wrow3 = W + (size_t)(n0 + 48 + r) * K;

  for (int k0 = 0; k0 < K; k0 += 32) {
    v16bf a0 = load_a_frag(arow0, k0, hi);
    v16bf a1 = load_a_frag(arow1, k0, hi);
    // B 32x16: lane r = column; halves = K ascending; lanes 16-31 carry K=16..31
    v16bf b0 = *(const v16bf*)(wrow0 + k0 + hi * 16);
    v16bf b1 = *(const v16bf*)(wrow1 + k0 + hi * 16);
    v16bf b2 = *(const v16bf*)(wrow2 + k0 + hi * 16);
    v16bf b3 = *(const v16bf*)(wrow3 + k0 + hi * 16);
    acc[0][0] = wmma_bf16(a0, b0, acc[0][0]);
    acc[1][0] = wmma_bf16(a1, b0, acc[1][0]);
    acc[0][1] = wmma_bf16(a0, b1, acc[0][1]);
    acc[1][1] = wmma_bf16(a1, b1, acc[1][1]);
    acc[0][2] = wmma_bf16(a0, b2, acc[0][2]);
    acc[1][2] = wmma_bf16(a1, b2, acc[1][2]);
    acc[0][3] = wmma_bf16(a0, b3, acc[0][3]);
    acc[1][3] = wmma_bf16(a1, b3, acc[1][3]);
  }

#pragma unroll
  for (int t = 0; t < 4; ++t) {
    const int nc = n0 + 16 * t + r;
    const float bv = bias[nc];
#pragma unroll
    for (int mi = 0; mi < 2; ++mi) {
#pragma unroll
      for (int j = 0; j < 8; ++j) {
        const int row = m0 + 16 * mi + j + 8 * hi;  // C layout: row = vgpr + 8*hi
        const float v = acc[mi][t][j] + bv;
        if (Cf) Cf[(size_t)row * N + nc] = v;
        else    Cb[(size_t)row * N + nc] = (bf16_t)v;
      }
    }
  }
}

// ---------------- RoPE + [b,s,h,d] -> [b,h,s,d] transpose ----------------
__global__ void rope_transpose(const bf16_t* __restrict__ in, const float* __restrict__ cosT,
                               const float* __restrict__ sinT, bf16_t* __restrict__ out,
                               int heads) {
  const int d = threadIdx.x;          // 0..127
  const int s = blockIdx.x;
  const int h = blockIdx.y;
  const int b = blockIdx.z;
  const size_t inBase = (((size_t)b * S + s) * heads + h) * D;
  const float x  = (float)in[inBase + d];
  const int   d2 = (d < 64) ? d + 64 : d - 64;
  const float x2 = (float)in[inBase + d2];
  const float rot = (d < 64) ? -x2 : x2;
  const float c  = cosT[s * D + d];
  const float sn = sinT[s * D + d];
  out[(((size_t)b * heads + h) * S + s) * D + d] = (bf16_t)(x * c + rot * sn);
}

// ---------------- V: [b,s,kv,d] -> [b,kv,d,s] ----------------
__global__ void transpose_v(const bf16_t* __restrict__ in, bf16_t* __restrict__ out) {
  const int d = threadIdx.x;
  const int s = blockIdx.x;
  const int h = blockIdx.y;
  const int b = blockIdx.z;
  out[(((size_t)b * KVH + h) * D + d) * S + s] =
      in[(((size_t)b * S + s) * KVH + h) * D + d];
}

// ---------------- causal GQA flash attention, one wave per 16-query tile ----------------
__global__ void __launch_bounds__(32)
gqa_attn(const bf16_t* __restrict__ Qr, const bf16_t* __restrict__ Kr,
         const bf16_t* __restrict__ Vt, bf16_t* __restrict__ Ctx) {
  __shared__ bf16_t Plds[16 * 32];   // P tile for C-layout -> A-layout transpose

  const int lane = threadIdx.x;
  const int r  = lane & 15;
  const int hi = lane >> 4;
  const int q0 = blockIdx.x * 16;
  const int h  = blockIdx.y;
  const int b  = blockIdx.z;
  const int kv = h / G;
  const float scale = 0.08838834764831845f;   // 1/sqrt(128)

  const bf16_t* qbase = Qr + (((size_t)b * NH + h) * S + q0) * D;
  const bf16_t* kbase = Kr + (((size_t)b * KVH + kv) * S) * (size_t)D;
  const bf16_t* vbase = Vt + (((size_t)b * KVH + kv) * D) * (size_t)S;

  // Q fragments: 16x128 = 4 A-frags over the D dimension
  v16bf qa[4];
  const bf16_t* qrow = qbase + (size_t)r * D;
#pragma unroll
  for (int kk = 0; kk < 4; ++kk) qa[kk] = load_a_frag(qrow, kk * 32, hi);

  v8f o[8];
#pragma unroll
  for (int nc = 0; nc < 8; ++nc)
#pragma unroll
    for (int i = 0; i < 8; ++i) o[nc][i] = 0.0f;

  float m_i[8], l_i[8];
#pragma unroll
  for (int j = 0; j < 8; ++j) { m_i[j] = NEG_INF; l_i[j] = 0.0f; }

  for (int j0 = 0; j0 < q0 + 16; j0 += 32) {   // causal: only tiles with keys <= q0+15
    v8f sc0, sc1;
#pragma unroll
    for (int i = 0; i < 8; ++i) { sc0[i] = 0.0f; sc1[i] = 0.0f; }

    const bf16_t* krow0 = kbase + (size_t)(j0 + r) * D + hi * 16;
    const bf16_t* krow1 = kbase + (size_t)(j0 + 16 + r) * D + hi * 16;
#pragma unroll
    for (int kk = 0; kk < 4; ++kk) {
      v16bf b0 = *(const v16bf*)(krow0 + kk * 32);
      sc0 = wmma_bf16(qa[kk], b0, sc0);
      v16bf b1 = *(const v16bf*)(krow1 + kk * 32);
      sc1 = wmma_bf16(qa[kk], b1, sc1);
    }

    // online softmax in C-layout (row M = j + 8*hi lives in vgpr j, col = lane r)
    float alpha[8];
#pragma unroll
    for (int j = 0; j < 8; ++j) {
      const int qi = q0 + j + 8 * hi;
      float s0 = sc0[j] * scale; if (j0 + r      > qi) s0 = NEG_INF;
      float s1 = sc1[j] * scale; if (j0 + 16 + r > qi) s1 = NEG_INF;
      float mx = fmaxf(s0, s1);
#pragma unroll
      for (int msk = 1; msk < 16; msk <<= 1) mx = fmaxf(mx, __shfl_xor(mx, msk, 32));
      const float mnew = fmaxf(m_i[j], mx);
      const float a  = __expf(m_i[j] - mnew);
      const float p0 = __expf(s0 - mnew);
      const float p1 = __expf(s1 - mnew);
      float rs = p0 + p1;
#pragma unroll
      for (int msk = 1; msk < 16; msk <<= 1) rs += __shfl_xor(rs, msk, 32);
      l_i[j] = l_i[j] * a + rs;
      m_i[j] = mnew;
      alpha[j] = a;
      const int M = j + 8 * hi;
      Plds[M * 32 + r]      = (bf16_t)p0;
      Plds[M * 32 + 16 + r] = (bf16_t)p1;
    }

#pragma unroll
    for (int nc = 0; nc < 8; ++nc)
#pragma unroll
      for (int j = 0; j < 8; ++j) o[nc][j] *= alpha[j];

    __syncthreads();

    // P (16x32) re-read in A-layout: row = r, keys hi*16 .. hi*16+15
    v16bf pa;
    {
      v8bf c0 = *(const v8bf*)&Plds[r * 32 + hi * 16];
      v8bf c1 = *(const v8bf*)&Plds[r * 32 + hi * 16 + 8];
#pragma unroll
      for (int i = 0; i < 8; ++i) { pa[i] = c0[i]; pa[i + 8] = c1[i]; }
    }

    // O += P @ V_tile, V as B-frags from [d][s] layout (contiguous per lane)
#pragma unroll
    for (int nc = 0; nc < 8; ++nc) {
      v16bf vb = *(const v16bf*)(vbase + (size_t)(nc * 16 + r) * S + j0 + hi * 16);
      o[nc] = wmma_bf16(pa, vb, o[nc]);
    }
    __syncthreads();
  }

  // finalize: divide by row sums, write context [b, s, h*D + d]
  bf16_t* cbase = Ctx + ((size_t)b * S + q0) * H + (size_t)h * D;
#pragma unroll
  for (int j = 0; j < 8; ++j) {
    const float inv = 1.0f / l_i[j];
    const int M = j + 8 * hi;
#pragma unroll
    for (int nc = 0; nc < 8; ++nc)
      cbase[(size_t)M * H + nc * 16 + r] = (bf16_t)(o[nc][j] * inv);
  }
}

// ---------------- host-side orchestration ----------------
extern "C" void kernel_launch(void* const* d_in, const int* in_sizes, int n_in,
                              void* d_out, int out_size, void* d_ws, size_t ws_size,
                              hipStream_t stream) {
  const float* X    = (const float*)d_in[0];
  const float* cosT = (const float*)d_in[1];
  const float* sinT = (const float*)d_in[2];
  const float* Wq   = (const float*)d_in[3];
  const float* bq   = (const float*)d_in[4];
  const float* Wk   = (const float*)d_in[5];
  const float* bk   = (const float*)d_in[6];
  const float* Wv   = (const float*)d_in[7];
  const float* bv   = (const float*)d_in[8];
  const float* Wo   = (const float*)d_in[9];
  const float* bo   = (const float*)d_in[10];
  float* out = (float*)d_out;

  char* ws = (char*)d_ws;
  // workspace layout (bytes)
  const size_t szXb  = (size_t)M_TOK * H * 2;        // 16 MB
  const size_t szWq  = (size_t)H * H * 2;            //  8 MB
  const size_t szWkv = (size_t)KVD * H * 2;          //  2 MB
  const size_t szQ   = (size_t)M_TOK * H * 2;        // 16 MB
  const size_t szKV  = (size_t)M_TOK * KVD * 2;      //  4 MB

  bf16_t* Xb   = (bf16_t*)(ws);
  bf16_t* Wqb  = (bf16_t*)(ws + szXb);
  bf16_t* Wkb  = (bf16_t*)(ws + szXb + szWq);
  bf16_t* Wvb  = (bf16_t*)(ws + szXb + szWq + szWkv);
  bf16_t* Wob  = (bf16_t*)(ws + szXb + szWq + 2 * szWkv);
  char*   p    = ws + szXb + 2 * szWq + 2 * szWkv;
  bf16_t* Qtmp = (bf16_t*)(p);                 p += szQ;   // reused as Ctx later
  bf16_t* Ktmp = (bf16_t*)(p);                 p += szKV;
  bf16_t* Vtmp = (bf16_t*)(p);                 p += szKV;
  bf16_t* Qr   = (bf16_t*)(p);                 p += szQ;
  bf16_t* Kr   = (bf16_t*)(p);                 p += szKV;
  bf16_t* Vt   = (bf16_t*)(p);
  bf16_t* Ctx  = Qtmp;   // Qtmp no longer needed once Qr exists

  // 1) fp32 -> bf16 conversions
  auto cvt = [&](const float* src, bf16_t* dst, int n) {
    cvt_f32_bf16<<<(n + 255) / 256, 256, 0, stream>>>(src, dst, n);
  };
  cvt(X,  Xb,  M_TOK * H);
  cvt(Wq, Wqb, H * H);
  cvt(Wk, Wkb, KVD * H);
  cvt(Wv, Wvb, KVD * H);
  cvt(Wo, Wob, H * H);

  // 2) QKV projections (bf16 out, fused bias); 32x64 wave tiles
  gemm_bf16_wave<<<dim3(H / 64,   M_TOK / 32), 32, 0, stream>>>(Xb, Wqb, bq, Qtmp, nullptr, M_TOK, H,   H);
  gemm_bf16_wave<<<dim3(KVD / 64, M_TOK / 32), 32, 0, stream>>>(Xb, Wkb, bk, Ktmp, nullptr, M_TOK, KVD, H);
  gemm_bf16_wave<<<dim3(KVD / 64, M_TOK / 32), 32, 0, stream>>>(Xb, Wvb, bv, Vtmp, nullptr, M_TOK, KVD, H);

  // 3) RoPE + head transposes
  rope_transpose<<<dim3(S, NH,  B), D, 0, stream>>>(Qtmp, cosT, sinT, Qr, NH);
  rope_transpose<<<dim3(S, KVH, B), D, 0, stream>>>(Ktmp, cosT, sinT, Kr, KVH);
  transpose_v  <<<dim3(S, KVH, B), D, 0, stream>>>(Vtmp, Vt);

  // 4) causal GQA flash attention -> Ctx (reuses Qtmp storage)
  gqa_attn<<<dim3(S / 16, NH, B), 32, 0, stream>>>(Qr, Kr, Vt, Ctx);

  // 5) output projection (fp32 out + bias) straight into d_out
  gemm_bf16_wave<<<dim3(H / 64, M_TOK / 32), 32, 0, stream>>>(Ctx, Wob, bo, nullptr, out, M_TOK, H, H);

  (void)in_sizes; (void)n_in; (void)out_size; (void)ws_size;
}
